// Embed_Hinge_87694642250038
// MI455X (gfx1250) — matrix operand order repaired and checked by reading, not compiled, not run on hardware
//
#include <hip/hip_runtime.h>
#include <hip/hip_bf16.h>
#include <math.h>

#define NN 100000
#define NE 800000
#define NG 512

typedef __attribute__((ext_vector_type(16))) _Float16 v16h;
typedef __attribute__((ext_vector_type(8)))  _Float16 v8h;
typedef __attribute__((ext_vector_type(8)))  float    v8f;

__device__ __forceinline__ v8f wmma16(v16h a, v16h b, v8f c) {
    return __builtin_amdgcn_wmma_f32_16x16x32_f16(
        /*neg_a=*/false, a, /*neg_b=*/false, b,
        /*c_mod=*/(short)0, c, /*reuse_a=*/false, /*reuse_b=*/false);
}

// A-matrix register image: elements 0..7 = p[0..7], elements 8..15 = p[16..23]
// (caller passes p = row + kbase + base8; see ISA 16-bit A 16x32 layout)
__device__ __forceinline__ v16h ldA(const _Float16* p) {
    v8h lo = *(const v8h*)p;
    v8h hi = *(const v8h*)(p + 16);
    v16h r;
#pragma unroll
    for (int i = 0; i < 8; ++i) { r[i] = lo[i]; r[i + 8] = hi[i]; }
    return r;
}
// B-matrix register image: elements 0..7 = p[0..7], elements 8..15 = p[16..23] (column-major weight slice)
__device__ __forceinline__ v16h ldB(const _Float16* p) {
    v8h lo = *(const v8h*)p;
    v8h hi = *(const v8h*)(p + 16);
    v16h r;
#pragma unroll
    for (int i = 0; i < 8; ++i) { r[i] = lo[i]; r[i + 8] = hi[i]; }
    return r;
}
__device__ __forceinline__ void lds_fence() {
    asm volatile("s_wait_dscnt 0" ::: "memory");
}

// ---------------- encoders (tiny GEMMs, scalar) ----------------
__global__ __launch_bounds__(256) void enc_node_kernel(
    const float* __restrict__ nf, const float* __restrict__ w,
    const float* __restrict__ b, float* __restrict__ h, _Float16* __restrict__ hh) {
    int n = blockIdx.x * 256 + threadIdx.x;
    if (n >= NN) return;
    float x[16];
#pragma unroll
    for (int k = 0; k < 16; ++k) x[k] = nf[(size_t)n * 16 + k];
    for (int c = 0; c < 32; ++c) {
        float a = b[c];
#pragma unroll
        for (int k = 0; k < 16; ++k) a += x[k] * w[k * 32 + c];
        h[(size_t)n * 32 + c]  = a;
        hh[(size_t)n * 32 + c] = (_Float16)a;
    }
}

__global__ __launch_bounds__(256) void enc_edge_kernel(
    const float* __restrict__ ef, const float* __restrict__ w,
    const float* __restrict__ b, _Float16* __restrict__ eh) {
    int e = blockIdx.x * 256 + threadIdx.x;
    if (e >= NE) return;
    float x[8];
#pragma unroll
    for (int k = 0; k < 8; ++k) x[k] = ef[(size_t)e * 8 + k];
    for (int c = 0; c < 16; ++c) {
        float a = b[c];
#pragma unroll
        for (int k = 0; k < 8; ++k) a += x[k] * w[k * 16 + c];
        eh[(size_t)e * 16 + c] = (_Float16)a;
    }
}

// ---------------- edge message MLP (WMMA) ----------------
// 8 waves/block, 16 edges/wave, both directions from one staged tile.
__global__ __launch_bounds__(256) void msg_kernel(
    const _Float16* __restrict__ hh, const _Float16* __restrict__ eh,
    const int* __restrict__ from_idx, const int* __restrict__ to_idx,
    const float* __restrict__ w1, const float* __restrict__ b1,
    const float* __restrict__ w2, const float* __restrict__ b2,
    float* __restrict__ agg) {
    __shared__ __align__(64) _Float16 sW1[32 * 96];     // [col][K padded 80->96]
    __shared__ __align__(64) _Float16 sW2[32 * 32];     // [col][K]
    __shared__ float sB1[32], sB2[32];
    __shared__ __align__(64) _Float16 sAct[8][16 * 96]; // per-wave 16x96 fp16
    __shared__ __align__(64) _Float16 sAct2[8][16 * 32];
    __shared__ int sIdx[8][2][16];                      // [wave][0=from,1=to][edge]

    const int tid = threadIdx.x;
    for (int i = tid; i < 32 * 96; i += 256) {
        int col = i / 96, k = i % 96;
        sW1[i] = (_Float16)((k < 80) ? w1[k * 32 + col] : 0.0f);
    }
    for (int i = tid; i < 32 * 32; i += 256) {
        int col = i >> 5, k = i & 31;
        sW2[i] = (_Float16)w2[k * 32 + col];
    }
    if (tid < 32) { sB1[tid] = b1[tid]; sB2[tid] = b2[tid]; }
    __syncthreads();

    const int wave = tid >> 5, lane = tid & 31;
    const int eb = (blockIdx.x * 8 + wave) * 16;
    const int j = lane >> 1, sub = lane & 1;
    _Float16* arow = &sAct[wave][j * 96];
    if (sub == 0) {
        int f = from_idx[eb + j];
        sIdx[wave][0][j] = f;
        const uint4* src = (const uint4*)(hh + (size_t)f * 32);
        uint4* dst = (uint4*)arow;
#pragma unroll
        for (int q = 0; q < 4; ++q) dst[q] = src[q];
        const uint4* es = (const uint4*)(eh + (size_t)(eb + j) * 16);
        uint4* ed = (uint4*)(arow + 64);
        ed[0] = es[0]; ed[1] = es[1];
    } else {
        int t = to_idx[eb + j];
        sIdx[wave][1][j] = t;
        const uint4* src = (const uint4*)(hh + (size_t)t * 32);
        uint4* dst = (uint4*)(arow + 32);
#pragma unroll
        for (int q = 0; q < 4; ++q) dst[q] = src[q];
        uint4 z = make_uint4(0, 0, 0, 0);
        uint4* pd = (uint4*)(arow + 80);
        pd[0] = z; pd[1] = z;
    }
    lds_fence();

    const int m = lane & 15;
    const int hi = lane >> 4;
    const int base8 = hi * 8;
    const int hi16 = hi * 16;
    const _Float16* myrow = &sAct[wave][m * 96];

#pragma unroll
    for (int d = 0; d < 2; ++d) {
        // K-tile base offsets into the staged activations:
        // fwd = [fs(0) | ts(32) | e(64)], rev = [ts(32) | fs(0) | e(64)]
        const int ka0 = d ? 32 : 0;
        const int ka1 = d ? 0 : 32;
        v8f acc[2] = {};
        {
            v16h A = ldA(myrow + ka0 + base8);
#pragma unroll
            for (int nt = 0; nt < 2; ++nt)
                acc[nt] = wmma16(A, ldB(&sW1[(m + 16 * nt) * 96 + 0 + base8]), acc[nt]);
        }
        {
            v16h A = ldA(myrow + ka1 + base8);
#pragma unroll
            for (int nt = 0; nt < 2; ++nt)
                acc[nt] = wmma16(A, ldB(&sW1[(m + 16 * nt) * 96 + 32 + base8]), acc[nt]);
        }
        {
            v16h A = ldA(myrow + 64 + base8);
#pragma unroll
            for (int nt = 0; nt < 2; ++nt)
                acc[nt] = wmma16(A, ldB(&sW1[(m + 16 * nt) * 96 + 64 + base8]), acc[nt]);
        }
        // bias + relu -> fp16 A-tile for layer 2
#pragma unroll
        for (int nt = 0; nt < 2; ++nt) {
            int col = m + 16 * nt;
            float bb = sB1[col];
#pragma unroll
            for (int r = 0; r < 8; ++r) {
                float v = acc[nt][r] + bb;
                sAct2[wave][(r + 8 * hi) * 32 + col] = (_Float16)(v > 0.f ? v : 0.f);
            }
        }
        lds_fence();
        v8f acc2[2] = {};
        {
            v16h A = ldA(&sAct2[wave][m * 32 + base8]);
#pragma unroll
            for (int nt = 0; nt < 2; ++nt)
                acc2[nt] = wmma16(A, ldB(&sW2[(m + 16 * nt) * 32 + base8]), acc2[nt]);
        }
        // scatter-add into agg (segment_sum): fwd -> to_idx, rev -> from_idx
        const int* dsts = sIdx[wave][d ? 0 : 1];
#pragma unroll
        for (int nt = 0; nt < 2; ++nt) {
            int col = m + 16 * nt;
            float bb = sB2[col];
#pragma unroll
            for (int r = 0; r < 8; ++r) {
                int dst = dsts[r + 8 * hi];
                atomicAdd(&agg[(size_t)dst * 32 + col], acc2[nt][r] + bb);
            }
        }
        lds_fence(); // WAR on sAct2 before next direction
    }
}

// ---------------- node update MLP (WMMA), h += MLP([agg,h]); zeroes agg ----------------
__global__ __launch_bounds__(64) void upd_kernel(
    const float* __restrict__ w1, const float* __restrict__ b1,
    const float* __restrict__ w2, const float* __restrict__ b2,
    float* __restrict__ h, _Float16* __restrict__ hh, float* __restrict__ agg) {
    __shared__ __align__(64) _Float16 sW1[64 * 64];   // [col][K]
    __shared__ __align__(64) _Float16 sW2[32 * 64];   // [col][K]
    __shared__ float sB1[64], sB2[32];
    __shared__ __align__(64) _Float16 sAct[2][16 * 64];
    __shared__ __align__(64) _Float16 sAct2[2][16 * 64];

    const int tid = threadIdx.x;
    for (int i = tid; i < 64 * 64; i += 64) {
        int col = i >> 6, k = i & 63;
        sW1[i] = (_Float16)w1[k * 64 + col];
    }
    for (int i = tid; i < 32 * 64; i += 64) {
        int col = i >> 6, k = i & 63;
        sW2[i] = (_Float16)w2[k * 32 + col];
    }
    if (tid < 64) sB1[tid] = b1[tid];
    if (tid < 32) sB2[tid] = b2[tid];
    __syncthreads();

    const int wave = tid >> 5, lane = tid & 31;
    const int nb = (blockIdx.x * 2 + wave) * 16;
    const int j = lane >> 1, sub = lane & 1;
    {
        int node = nb + j;
        float* src = sub ? (h + (size_t)node * 32) : (agg + (size_t)node * 32);
        _Float16* dst = &sAct[wave][j * 64 + sub * 32];
#pragma unroll
        for (int q = 0; q < 32; ++q) dst[q] = (_Float16)src[q];
        if (!sub) {
#pragma unroll
            for (int q = 0; q < 32; ++q) src[q] = 0.f;  // re-zero agg for next round
        }
    }
    lds_fence();

    const int m = lane & 15, hi = lane >> 4, base8 = hi * 8, hi16 = hi * 16;
    const _Float16* myrow = &sAct[wave][m * 64];
    v8f acc[4] = {};
#pragma unroll
    for (int kt = 0; kt < 2; ++kt) {
        v16h A = ldA(myrow + kt * 32 + base8);
#pragma unroll
        for (int nt = 0; nt < 4; ++nt)
            acc[nt] = wmma16(A, ldB(&sW1[(m + 16 * nt) * 64 + kt * 32 + base8]), acc[nt]);
    }
#pragma unroll
    for (int nt = 0; nt < 4; ++nt) {
        int col = m + 16 * nt;
        float bb = sB1[col];
#pragma unroll
        for (int r = 0; r < 8; ++r) {
            float v = acc[nt][r] + bb;
            sAct2[wave][(r + 8 * hi) * 64 + col] = (_Float16)(v > 0.f ? v : 0.f);
        }
    }
    lds_fence();
    v8f acc2[2] = {};
    const _Float16* myrow2 = &sAct2[wave][m * 64];
#pragma unroll
    for (int kt = 0; kt < 2; ++kt) {
        v16h A = ldA(myrow2 + kt * 32 + base8);
#pragma unroll
        for (int nt = 0; nt < 2; ++nt)
            acc2[nt] = wmma16(A, ldB(&sW2[(m + 16 * nt) * 64 + kt * 32 + base8]), acc2[nt]);
    }
#pragma unroll
    for (int nt = 0; nt < 2; ++nt) {
        int col = m + 16 * nt;
        float bb = sB2[col];
#pragma unroll
        for (int r = 0; r < 8; ++r) {
            size_t off = (size_t)(nb + r + 8 * hi) * 32 + col;
            float nv = h[off] + acc2[nt][r] + bb;
            h[off]  = nv;
            hh[off] = (_Float16)nv;
        }
    }
}

// ---------------- gated graph aggregation (WMMA 32->128 + sigmoid gate + scatter) ----------------
__global__ __launch_bounds__(64) void gate_kernel(
    const _Float16* __restrict__ hh, const int* __restrict__ graph_idx,
    const float* __restrict__ w1, const float* __restrict__ b1,
    float* __restrict__ gstates) {
    __shared__ __align__(64) _Float16 sW1[128 * 32];  // [col][K]
    __shared__ float sB1[128];
    __shared__ __align__(64) _Float16 sAct[2][16 * 32];

    const int tid = threadIdx.x;
    for (int i = tid; i < 128 * 32; i += 64) {
        int col = i >> 5, k = i & 31;
        sW1[i] = (_Float16)w1[k * 128 + col];
    }
    if (tid < 64) { sB1[tid] = b1[tid]; sB1[tid + 64] = b1[tid + 64]; }
    __syncthreads();

    const int wave = tid >> 5, lane = tid & 31;
    const int nb = (blockIdx.x * 2 + wave) * 16;
    const int j = lane >> 1, sub = lane & 1;
    {
        const uint4* src = (const uint4*)(hh + (size_t)(nb + j) * 32 + sub * 16);
        uint4* dst = (uint4*)(&sAct[wave][j * 32 + sub * 16]);
        dst[0] = src[0]; dst[1] = src[1];
    }
    lds_fence();
    const int m = lane & 15, hi = lane >> 4, base8 = hi * 8, hi16 = hi * 16;
    v16h A = ldA(&sAct[wave][m * 32 + base8]);
    v8f acc[8] = {};
#pragma unroll
    for (int nt = 0; nt < 8; ++nt)
        acc[nt] = wmma16(A, ldB(&sW1[(m + 16 * nt) * 32 + base8]), acc[nt]);
#pragma unroll
    for (int t = 0; t < 4; ++t) {
        int col = m + 16 * t;
        float ba = sB1[col], bg = sB1[col + 64];
#pragma unroll
        for (int r = 0; r < 8; ++r) {
            int node = nb + r + 8 * hi;
            float a  = acc[t][r] + ba;
            float bv = acc[t + 4][r] + bg;
            float gated = bv / (1.f + __expf(-a));
            int g = graph_idx[node];
            atomicAdd(&gstates[(size_t)g * 64 + col], gated);
        }
    }
}

// ---------------- final projection + L1 hinge (fp32 scalar) ----------------
__global__ __launch_bounds__(256) void final_kernel(
    const float* __restrict__ gs, const float* __restrict__ w,
    const float* __restrict__ b, float* __restrict__ out) {
    int i = threadIdx.x;  // 0..255, one graph pair each
    const float* xr = gs + (size_t)(2 * i) * 64;
    const float* yr = gs + (size_t)(2 * i + 1) * 64;
    float acc = 0.f;
    for (int c = 0; c < 64; ++c) {
        float xv = b[c], yv = b[c];
        for (int k = 0; k < 64; ++k) {
            float wv = w[k * 64 + c];
            xv += xr[k] * wv;
            yv += yr[k] * wv;
        }
        acc += fabsf(yv - xv);  // relu(y-x)^1 + relu(x-y)^1
    }
    out[i] = acc;
}

extern "C" void kernel_launch(void* const* d_in, const int* in_sizes, int n_in,
                              void* d_out, int out_size, void* d_ws, size_t ws_size,
                              hipStream_t stream) {
    const float* nf   = (const float*)d_in[0];
    const float* ef   = (const float*)d_in[1];
    const int* fidx   = (const int*)d_in[2];
    const int* tidx   = (const int*)d_in[3];
    const int* gidx   = (const int*)d_in[4];
    const float* enw  = (const float*)d_in[5];
    const float* enb  = (const float*)d_in[6];
    const float* eew  = (const float*)d_in[7];
    const float* eeb  = (const float*)d_in[8];
    const float* mw1  = (const float*)d_in[9];
    const float* mb1  = (const float*)d_in[10];
    const float* mw2  = (const float*)d_in[11];
    const float* mb2  = (const float*)d_in[12];
    const float* uw1  = (const float*)d_in[13];
    const float* ub1  = (const float*)d_in[14];
    const float* uw2  = (const float*)d_in[15];
    const float* ub2  = (const float*)d_in[16];
    const float* aw1  = (const float*)d_in[17];
    const float* ab1  = (const float*)d_in[18];
    const float* aw2  = (const float*)d_in[19];
    const float* ab2  = (const float*)d_in[20];

    char* ws = (char*)d_ws;
    float*    h   = (float*)   (ws + 0);           // 100000*32 f32 = 12.8 MB
    float*    agg = (float*)   (ws + 12800000);    // 100000*32 f32 = 12.8 MB
    _Float16* hhv = (_Float16*)(ws + 25600000);    // 100000*32 f16 =  6.4 MB
    _Float16* ehv = (_Float16*)(ws + 32000000);    // 800000*16 f16 = 25.6 MB
    float*    gst = (float*)   (ws + 57600000);    // 512*64 f32

    hipMemsetAsync(agg, 0, (size_t)NN * 32 * sizeof(float), stream);
    hipMemsetAsync(gst, 0, (size_t)NG * 64 * sizeof(float), stream);

    enc_node_kernel<<<(NN + 255) / 256, 256, 0, stream>>>(nf, enw, enb, h, hhv);
    enc_edge_kernel<<<NE / 256, 256, 0, stream>>>(ef, eew, eeb, ehv);

    for (int p = 0; p < 3; ++p) {
        msg_kernel<<<NE / 128, 256, 0, stream>>>(hhv, ehv, fidx, tidx,
                                                 mw1, mb1, mw2, mb2, agg);
        upd_kernel<<<NN / 32, 64, 0, stream>>>(uw1, ub1, uw2, ub2, h, hhv, agg);
    }
    gate_kernel<<<NN / 32, 64, 0, stream>>>(hhv, gidx, aw1, ab1, gst);
    final_kernel<<<1, 256, 0, stream>>>(gst, aw2, ab2, (float*)d_out);
}
